// PointNet_Aux_48610439856845
// MI455X (gfx1250) — matrix operand order repaired and checked by reading, not compiled, not run on hardware
//
#include <hip/hip_runtime.h>
#include <hip/hip_bf16.h>

// ---------------------------------------------------------------------------
// PointNet++ (SSG) forward for MI455X / gfx1250.
// All MLP GEMMs use V_WMMA_F32_16X16X32_F16 (f16 A/B, f32 accum).
// ---------------------------------------------------------------------------

typedef __attribute__((ext_vector_type(16))) _Float16 v16h;
typedef __attribute__((ext_vector_type(8)))  float    v8f;

#define NB 8          // batch
#define NPTS 2048     // input points
#define CIN 7
#define M1 1024
#define M2 256
#define KNB 64        // neighbors per ball
#define NEG_INF_KEY 0x007FFFFFu

// monotonic float <-> uint key (total order, works for atomicMax)
__device__ __forceinline__ unsigned f2key(float x) {
  unsigned u = __float_as_uint(x);
  return (u & 0x80000000u) ? ~u : (u | 0x80000000u);
}
__device__ __forceinline__ float key2f(unsigned k) {
  return (k & 0x80000000u) ? __uint_as_float(k & 0x7FFFFFFFu)
                           : __uint_as_float(~k);
}

// ---------------- WMMA fragment loaders (CDNA5 ISA layouts) ----------------
// A: 16x32 f16 tile. lanes 0-15: M=lane, K blocks {0..7,16..23};
//                    lanes 16-31: M=lane-16, K blocks {8..15,24..31}.
__device__ __forceinline__ v16h load_a_frag(const _Float16* base, int lda, int lane) {
  int row = lane & 15;
  int kb  = (lane >> 4) << 3;   // 0 or 8
  const _Float16* p = base + row * lda + kb;
  v16h a;
#pragma unroll
  for (int i = 0; i < 8; ++i) a[i] = p[i];
#pragma unroll
  for (int i = 0; i < 8; ++i) a[8 + i] = p[16 + i];
  return a;
}
// B: 32x16 f16 tile. N=lane%16; lanes 0-15 K {0..7,16..23}, lanes 16-31 K {8..15,24..31}.
__device__ __forceinline__ v16h load_b_frag(const _Float16* base, int ldb, int lane) {
  int kb = (lane >> 4) << 3;
  const _Float16* p = base + (lane & 15) + kb * ldb;
  v16h b;
#pragma unroll
  for (int i = 0; i < 8; ++i) b[i] = p[i * ldb];
#pragma unroll
  for (int i = 0; i < 8; ++i) b[8 + i] = p[(16 + i) * ldb];
  return b;
}

// ------------- generic fused GEMM layers (block = 128 thr = 4 waves) -------
// out = relu((A @ W + bias) * sc + sh), A: ROWSxKD (LDS f16), W: KDxCOLS (global f16)
template <int ROWS, int COLS, int KD, int NW>
__device__ __forceinline__ void gemm_relu_affine(
    const _Float16* A, int lda, const _Float16* W,
    const float* bias, const float* sc, const float* sh,
    _Float16* out, int ldo) {
  const int lane = threadIdx.x & 31;
  const int wv   = threadIdx.x >> 5;
  const int CT = COLS / 16, RT = ROWS / 16;
  for (int t = wv; t < RT * CT; t += NW) {
    int tm = t / CT, tn = t % CT;
    v8f acc = {};
#pragma unroll
    for (int k0 = 0; k0 < KD; k0 += 32) {
      v16h a = load_a_frag(A + (tm * 16) * lda + k0, lda, lane);
      v16h b = load_b_frag(W + k0 * COLS + tn * 16, COLS, lane);
      acc = __builtin_amdgcn_wmma_f32_16x16x32_f16(false, a, false, b,
                                                   (short)0, acc, false, false);
    }
    int n  = tn * 16 + (lane & 15);
    int r0 = tm * 16 + ((lane >> 4) << 3);
    float bn = bias[n], sn = sc[n], hn = sh[n];
#pragma unroll
    for (int v = 0; v < 8; ++v) {
      float e = (acc[v] + bn) * sn + hn;
      e = e > 0.f ? e : 0.f;
      out[(r0 + v) * ldo + n] = (_Float16)e;
    }
  }
}

// last layer: D = A @ W + bias, fused column-max over all ROWS into chmax keys
template <int ROWS, int COLS, int KD, int NW>
__device__ __forceinline__ void gemm_bias_colmax(
    const _Float16* A, int lda, const _Float16* W,
    const float* bias, unsigned* chmax) {
  const int lane = threadIdx.x & 31;
  const int wv   = threadIdx.x >> 5;
  const int CT = COLS / 16, RT = ROWS / 16;
  for (int t = wv; t < RT * CT; t += NW) {
    int tm = t / CT, tn = t % CT;
    v8f acc = {};
#pragma unroll
    for (int k0 = 0; k0 < KD; k0 += 32) {
      v16h a = load_a_frag(A + (tm * 16) * lda + k0, lda, lane);
      v16h b = load_b_frag(W + k0 * COLS + tn * 16, COLS, lane);
      acc = __builtin_amdgcn_wmma_f32_16x16x32_f16(false, a, false, b,
                                                   (short)0, acc, false, false);
    }
    int n = tn * 16 + (lane & 15);
    float bn = bias[n];
    float mx = -__builtin_inff();
#pragma unroll
    for (int v = 0; v < 8; ++v) mx = fmaxf(mx, acc[v] + bn);
    float other = __shfl_xor(mx, 16, 32);   // rows 0-7 <-> rows 8-15, same column
    mx = fmaxf(mx, other);
    if (lane < 16) atomicMax(&chmax[n], f2key(mx));
  }
}

// --------------------------- weight convert/pad ----------------------------
__global__ void cvt_w_kernel(const float* __restrict__ W, _Float16* __restrict__ out,
                             int din, int dout, int dinPad) {
  int i = blockIdx.x * 256 + threadIdx.x;
  if (i >= dinPad * dout) return;
  int k = i / dout, n = i - k * dout;
  out[i] = (k < din) ? (_Float16)W[k * dout + n] : (_Float16)0.0f;
}

__global__ void init_gkeys_kernel(unsigned* g, int n) {
  int i = blockIdx.x * 256 + threadIdx.x;
  if (i < n) g[i] = NEG_INF_KEY;
}

// ------------------------------- FPS ---------------------------------------
// one block per batch; sequential farthest-point sampling, dists in LDS.
__global__ __launch_bounds__(256) void fps_kernel(
    const float* __restrict__ pos, int rowStride, int batchStride,
    int N, int m, int* __restrict__ outIdx, float* __restrict__ outQ) {
  __shared__ float dists[NPTS];
  __shared__ float rd[256];
  __shared__ int   ri[256];
  __shared__ float curp[3];
  int b = blockIdx.x, tid = threadIdx.x;
  const float* P = pos + (size_t)b * batchStride;
  float p0x = P[0], p0y = P[1], p0z = P[2];
  for (int j = tid; j < N; j += 256) {
    float dx = P[j * rowStride + 0] - p0x;
    float dy = P[j * rowStride + 1] - p0y;
    float dz = P[j * rowStride + 2] - p0z;
    dists[j] = dx * dx + dy * dy + dz * dz;
  }
  if (tid == 0) {
    outIdx[b * m] = 0;
    outQ[(size_t)(b * m) * 3 + 0] = p0x;
    outQ[(size_t)(b * m) * 3 + 1] = p0y;
    outQ[(size_t)(b * m) * 3 + 2] = p0z;
  }
  __syncthreads();
  for (int t = 1; t < m; ++t) {
    float bd = -1.f; int bi = 0x7fffffff;
    for (int j = tid; j < N; j += 256) {
      float d = dists[j];
      if (d > bd) { bd = d; bi = j; }
    }
    rd[tid] = bd; ri[tid] = bi;
    __syncthreads();
    for (int s = 128; s > 0; s >>= 1) {
      if (tid < s) {
        float od = rd[tid + s]; int oi = ri[tid + s];
        if (od > rd[tid] || (od == rd[tid] && oi < ri[tid])) { rd[tid] = od; ri[tid] = oi; }
      }
      __syncthreads();
    }
    if (tid == 0) {
      int i = ri[0];
      float cx = P[i * rowStride + 0], cy = P[i * rowStride + 1], cz = P[i * rowStride + 2];
      curp[0] = cx; curp[1] = cy; curp[2] = cz;
      outIdx[b * m + t] = i;
      outQ[(size_t)(b * m + t) * 3 + 0] = cx;
      outQ[(size_t)(b * m + t) * 3 + 1] = cy;
      outQ[(size_t)(b * m + t) * 3 + 2] = cz;
    }
    __syncthreads();
    float cx = curp[0], cy = curp[1], cz = curp[2];
    for (int j = tid; j < N; j += 256) {
      float dx = P[j * rowStride + 0] - cx;
      float dy = P[j * rowStride + 1] - cy;
      float dz = P[j * rowStride + 2] - cz;
      float nd = dx * dx + dy * dy + dz * dz;
      if (nd < dists[j]) dists[j] = nd;
    }
    __syncthreads();
  }
}

// ---------------------------- ball query -----------------------------------
// one wave per query; compact in-radius candidates, select nearest K=64.
// invalid slots padded with the centroid's own index (valid, d2=0) so the
// downstream maxpool needs no mask (max over multiset == max over set).
#define CAND 512
__global__ __launch_bounds__(256) void ball_query_kernel(
    const float* __restrict__ pos, int rowStride, int batchStride, int Nsrc,
    const float* __restrict__ q, const int* __restrict__ fpsIdx,
    int M, float r2, int* __restrict__ nidx) {
  __shared__ float cd[8][CAND];
  __shared__ int   ci[8][CAND];
  __shared__ int   cnt[8];
  int w = threadIdx.x >> 5, lane = threadIdx.x & 31;
  int qi = blockIdx.x * 8 + w;
  int b = qi / M;
  const float* P = pos + (size_t)b * batchStride;
  float qx = q[qi * 3 + 0], qy = q[qi * 3 + 1], qz = q[qi * 3 + 2];
  if (lane == 0) cnt[w] = 0;
  asm volatile("s_wait_dscnt 0" ::: "memory");
  for (int j = lane; j < Nsrc; j += 32) {
    float dx = P[j * rowStride + 0] - qx;
    float dy = P[j * rowStride + 1] - qy;
    float dz = P[j * rowStride + 2] - qz;
    float d2 = dx * dx + dy * dy + dz * dz;
    if (d2 <= r2) {
      int s = atomicAdd(&cnt[w], 1);
      if (s < CAND) { cd[w][s] = d2; ci[w][s] = j; }
    }
  }
  asm volatile("s_wait_dscnt 0" ::: "memory");
  int C = cnt[w]; if (C > CAND) C = CAND;
  int* out = nidx + (size_t)qi * KNB;
  if (C <= KNB) {
    int pad = fpsIdx[qi];
    for (int k = lane; k < KNB; k += 32) out[k] = (k < C) ? ci[w][k] : pad;
  } else {
    for (int k = 0; k < KNB; ++k) {
      float bd = __builtin_inff(); int bp = -1;
      for (int c = lane; c < C; c += 32) {
        float d = cd[w][c];
        if (d < bd) { bd = d; bp = c; }
      }
#pragma unroll
      for (int off = 16; off > 0; off >>= 1) {
        float od = __shfl_xor(bd, off, 32);
        int   op = __shfl_xor(bp, off, 32);
        if (od < bd || (od == bd && (unsigned)op < (unsigned)bp)) { bd = od; bp = op; }
      }
      if (lane == 0) { out[k] = ci[w][bp]; cd[w][bp] = __builtin_inff(); }
      asm volatile("s_wait_dscnt 0" ::: "memory");
    }
  }
}

// --------------------------- SA stage 1 ------------------------------------
// one group (64 neighbors, 10->64->64->128 MLP + maxpool) per 128-thread block
__global__ __launch_bounds__(128) void sa_stage1_kernel(
    const float* __restrict__ data, const int* __restrict__ nidx,
    const float* __restrict__ q,
    const _Float16* __restrict__ w1, const _Float16* __restrict__ w2,
    const _Float16* __restrict__ w3,
    const float* __restrict__ b1, const float* __restrict__ s1, const float* __restrict__ h1,
    const float* __restrict__ b2, const float* __restrict__ s2, const float* __restrict__ h2,
    const float* __restrict__ b3,
    _Float16* __restrict__ x1out) {
  __shared__ _Float16 msg[64 * 32];
  __shared__ _Float16 hA[64 * 64];
  __shared__ _Float16 hB[64 * 64];
  __shared__ unsigned chmax[128];
  int g = blockIdx.x;
  int b = g >> 10;
  int tid = threadIdx.x;
  if (tid == 0) { __builtin_prefetch(w1, 0, 0); __builtin_prefetch(w2, 0, 0); __builtin_prefetch(w3, 0, 0); }
  float qx = q[g * 3 + 0], qy = q[g * 3 + 1], qz = q[g * 3 + 2];
  for (int e = tid; e < 64 * 32; e += 128) {
    int row = e >> 5, ch = e & 31;
    int j = nidx[(size_t)g * KNB + row];
    const float* pj = data + (size_t)(b * NPTS + j) * CIN;
    float v;
    if (ch < 7)       v = pj[ch];
    else if (ch == 7) v = pj[0] - qx;
    else if (ch == 8) v = pj[1] - qy;
    else if (ch == 9) v = pj[2] - qz;
    else              v = 0.f;
    msg[row * 32 + ch] = (_Float16)v;
  }
  if (tid < 128) chmax[tid] = NEG_INF_KEY;
  __syncthreads();
  gemm_relu_affine<64, 64, 32, 4>(msg, 32, w1, b1, s1, h1, hA, 64);
  __syncthreads();
  gemm_relu_affine<64, 64, 64, 4>(hA, 64, w2, b2, s2, h2, hB, 64);
  __syncthreads();
  gemm_bias_colmax<64, 128, 64, 4>(hB, 64, w3, b3, chmax);
  __syncthreads();
  if (tid < 128) x1out[(size_t)g * 128 + tid] = (_Float16)key2f(chmax[tid]);
}

// --------------------------- SA stage 2 ------------------------------------
// 131 channels padded to 160; 160->128->128->256 MLP + maxpool
__global__ __launch_bounds__(128) void sa_stage2_kernel(
    const _Float16* __restrict__ x1, const float* __restrict__ pos1,
    const float* __restrict__ q2, const int* __restrict__ nidx,
    const _Float16* __restrict__ w1, const _Float16* __restrict__ w2,
    const _Float16* __restrict__ w3,
    const float* __restrict__ b1, const float* __restrict__ s1, const float* __restrict__ h1,
    const float* __restrict__ b2, const float* __restrict__ s2, const float* __restrict__ h2,
    const float* __restrict__ b3,
    _Float16* __restrict__ x2out) {
  __shared__ _Float16 msg[64 * 160];
  __shared__ _Float16 hA[64 * 128];
  __shared__ _Float16 hB[64 * 128];
  __shared__ unsigned chmax[256];
  int g = blockIdx.x;
  int b = g >> 8;
  int tid = threadIdx.x;
  if (tid == 0) { __builtin_prefetch(w1, 0, 0); __builtin_prefetch(w2, 0, 0); __builtin_prefetch(w3, 0, 0); }
  float qx = q2[g * 3 + 0], qy = q2[g * 3 + 1], qz = q2[g * 3 + 2];
  for (int e = tid; e < 64 * 160; e += 128) {
    int row = e / 160, ch = e - row * 160;
    int j = nidx[(size_t)g * KNB + row];
    float v;
    if (ch < 128)       v = (float)x1[(size_t)(b * M1 + j) * 128 + ch];
    else if (ch == 128) v = pos1[(size_t)(b * M1 + j) * 3 + 0] - qx;
    else if (ch == 129) v = pos1[(size_t)(b * M1 + j) * 3 + 1] - qy;
    else if (ch == 130) v = pos1[(size_t)(b * M1 + j) * 3 + 2] - qz;
    else                v = 0.f;
    msg[row * 160 + ch] = (_Float16)v;
  }
  for (int e = tid; e < 256; e += 128) chmax[e] = NEG_INF_KEY;
  __syncthreads();
  gemm_relu_affine<64, 128, 160, 4>(msg, 160, w1, b1, s1, h1, hA, 128);
  __syncthreads();
  gemm_relu_affine<64, 128, 128, 4>(hA, 128, w2, b2, s2, h2, hB, 128);
  __syncthreads();
  gemm_bias_colmax<64, 256, 128, 4>(hB, 128, w3, b3, chmax);
  __syncthreads();
  for (int e = tid; e < 256; e += 128) x2out[(size_t)g * 256 + e] = (_Float16)key2f(chmax[e]);
}

// --------------------------- SA stage 3 ------------------------------------
// per-point MLP 259(pad 288)->256->512->1024, global max -> gkeys (per batch)
__global__ __launch_bounds__(128) void sa_stage3_kernel(
    const _Float16* __restrict__ x2, const float* __restrict__ q2,
    const _Float16* __restrict__ w1, const _Float16* __restrict__ w2,
    const _Float16* __restrict__ w3,
    const float* __restrict__ b1, const float* __restrict__ s1, const float* __restrict__ h1,
    const float* __restrict__ b2, const float* __restrict__ s2, const float* __restrict__ h2,
    const float* __restrict__ b3,
    unsigned* __restrict__ gkeys) {
  __shared__ _Float16 msg[16 * 288];
  __shared__ _Float16 hA[16 * 256];
  __shared__ _Float16 hB[16 * 512];
  __shared__ unsigned chmax[1024];
  int p0 = blockIdx.x * 16;   // global point base among NB*M2
  int b  = p0 >> 8;           // 256 points per batch
  int tid = threadIdx.x;
  if (tid == 0) { __builtin_prefetch(w1, 0, 0); __builtin_prefetch(w2, 0, 0); __builtin_prefetch(w3, 0, 0); }
  for (int e = tid; e < 16 * 288; e += 128) {
    int row = e / 288, ch = e - row * 288;
    int p = p0 + row;
    float v;
    if (ch < 256)      v = (float)x2[(size_t)p * 256 + ch];
    else if (ch < 259) v = q2[(size_t)p * 3 + (ch - 256)];
    else               v = 0.f;
    msg[row * 288 + ch] = (_Float16)v;
  }
  for (int e = tid; e < 1024; e += 128) chmax[e] = NEG_INF_KEY;
  __syncthreads();
  gemm_relu_affine<16, 256, 288, 4>(msg, 288, w1, b1, s1, h1, hA, 256);
  __syncthreads();
  gemm_relu_affine<16, 512, 256, 4>(hA, 256, w2, b2, s2, h2, hB, 512);
  __syncthreads();
  gemm_bias_colmax<16, 1024, 512, 4>(hB, 512, w3, b3, chmax);
  __syncthreads();
  for (int e = tid; e < 1024; e += 128) atomicMax(&gkeys[(size_t)b * 1024 + e], chmax[e]);
}

// ------------------------------- head --------------------------------------
__global__ __launch_bounds__(512) void head_kernel(
    const unsigned* __restrict__ gkeys,
    const float* __restrict__ W0, const float* __restrict__ c0,
    const float* __restrict__ W1, const float* __restrict__ c1,
    const float* __restrict__ W2, const float* __restrict__ c2,
    float* __restrict__ out) {
  __shared__ float gb[1024];
  __shared__ float y1[512];
  __shared__ float y2[256];
  int b = blockIdx.x, tid = threadIdx.x;
  for (int e = tid; e < 1024; e += 512) gb[e] = key2f(gkeys[(size_t)b * 1024 + e]);
  __syncthreads();
  {
    float acc = c0[tid];
    for (int k = 0; k < 1024; ++k) acc += gb[k] * W0[k * 512 + tid];
    y1[tid] = fmaxf(acc, 0.f);
  }
  __syncthreads();
  if (tid < 256) {
    float acc = c1[tid];
    for (int k = 0; k < 512; ++k) acc += y1[k] * W1[k * 256 + tid];
    y2[tid] = fmaxf(acc, 0.f);
  }
  __syncthreads();
  if (tid < 49) {
    float acc = c2[tid];
    for (int k = 0; k < 256; ++k) acc += y2[k] * W2[k * 49 + tid];
    out[b * 49 + tid] = acc;
  }
}

// ------------------------------- host --------------------------------------
extern "C" void kernel_launch(void* const* d_in, const int* in_sizes, int n_in,
                              void* d_out, int out_size, void* d_ws, size_t ws_size,
                              hipStream_t stream) {
  (void)in_sizes; (void)n_in; (void)out_size; (void)ws_size;
  const float* data = (const float*)d_in[0];
  const float* sa1W[3] = {(const float*)d_in[1], (const float*)d_in[2], (const float*)d_in[3]};
  const float* sa1b[3] = {(const float*)d_in[4], (const float*)d_in[5], (const float*)d_in[6]};
  const float* sa1s[2] = {(const float*)d_in[7], (const float*)d_in[8]};
  const float* sa1h[2] = {(const float*)d_in[9], (const float*)d_in[10]};
  const float* sa2W[3] = {(const float*)d_in[11], (const float*)d_in[12], (const float*)d_in[13]};
  const float* sa2b[3] = {(const float*)d_in[14], (const float*)d_in[15], (const float*)d_in[16]};
  const float* sa2s[2] = {(const float*)d_in[17], (const float*)d_in[18]};
  const float* sa2h[2] = {(const float*)d_in[19], (const float*)d_in[20]};
  const float* sa3W[3] = {(const float*)d_in[21], (const float*)d_in[22], (const float*)d_in[23]};
  const float* sa3b[3] = {(const float*)d_in[24], (const float*)d_in[25], (const float*)d_in[26]};
  const float* sa3s[2] = {(const float*)d_in[27], (const float*)d_in[28]};
  const float* sa3h[2] = {(const float*)d_in[29], (const float*)d_in[30]};
  const float* hW[3]   = {(const float*)d_in[31], (const float*)d_in[32], (const float*)d_in[33]};
  const float* hb[3]   = {(const float*)d_in[34], (const float*)d_in[35], (const float*)d_in[36]};
  float* out = (float*)d_out;

  // workspace carve-out (256B aligned)
  char* ws = (char*)d_ws;
  size_t off = 0;
  auto take = [&](size_t bytes) -> void* {
    off = (off + 255) & ~(size_t)255;
    void* p = ws + off;
    off += bytes;
    return p;
  };
  _Float16* w1a = (_Float16*)take(32  * 64   * 2);
  _Float16* w2a = (_Float16*)take(64  * 64   * 2);
  _Float16* w3a = (_Float16*)take(64  * 128  * 2);
  _Float16* w1b = (_Float16*)take(160 * 128  * 2);
  _Float16* w2b = (_Float16*)take(128 * 128  * 2);
  _Float16* w3b = (_Float16*)take(128 * 256  * 2);
  _Float16* w1c = (_Float16*)take(288 * 256  * 2);
  _Float16* w2c = (_Float16*)take(256 * 512  * 2);
  _Float16* w3c = (_Float16*)take(512 * 1024 * 2);
  int*      fps1  = (int*)take(NB * M1 * 4);
  float*    q1    = (float*)take((size_t)NB * M1 * 3 * 4);
  int*      nidx1 = (int*)take((size_t)NB * M1 * KNB * 4);
  _Float16* x1    = (_Float16*)take((size_t)NB * M1 * 128 * 2);
  int*      fps2  = (int*)take(NB * M2 * 4);
  float*    q2    = (float*)take((size_t)NB * M2 * 3 * 4);
  int*      nidx2 = (int*)take((size_t)NB * M2 * KNB * 4);
  _Float16* x2    = (_Float16*)take((size_t)NB * M2 * 256 * 2);
  unsigned* gkeys = (unsigned*)take(NB * 1024 * 4);

  // 1) convert + pad weights to f16
  auto cvt = [&](const float* W, _Float16* dst, int din, int dout, int dinPad) {
    int n = dinPad * dout;
    cvt_w_kernel<<<(n + 255) / 256, 256, 0, stream>>>(W, dst, din, dout, dinPad);
  };
  cvt(sa1W[0], w1a, 10, 64, 32);   cvt(sa1W[1], w2a, 64, 64, 64);   cvt(sa1W[2], w3a, 64, 128, 64);
  cvt(sa2W[0], w1b, 131, 128, 160);cvt(sa2W[1], w2b, 128, 128, 128);cvt(sa2W[2], w3b, 128, 256, 128);
  cvt(sa3W[0], w1c, 259, 256, 288);cvt(sa3W[1], w2c, 256, 512, 256);cvt(sa3W[2], w3c, 512, 1024, 512);
  init_gkeys_kernel<<<(NB * 1024 + 255) / 256, 256, 0, stream>>>(gkeys, NB * 1024);

  // 2) SA1: FPS -> ball query -> fused WMMA MLP + maxpool
  fps_kernel<<<NB, 256, 0, stream>>>(data, CIN, NPTS * CIN, NPTS, M1, fps1, q1);
  ball_query_kernel<<<(NB * M1) / 8, 256, 0, stream>>>(data, CIN, NPTS * CIN, NPTS,
                                                       q1, fps1, M1, 0.2f * 0.2f, nidx1);
  sa_stage1_kernel<<<NB * M1, 128, 0, stream>>>(data, nidx1, q1, w1a, w2a, w3a,
                                                sa1b[0], sa1s[0], sa1h[0],
                                                sa1b[1], sa1s[1], sa1h[1],
                                                sa1b[2], x1);

  // 3) SA2
  fps_kernel<<<NB, 256, 0, stream>>>(q1, 3, M1 * 3, M1, M2, fps2, q2);
  ball_query_kernel<<<(NB * M2) / 8, 256, 0, stream>>>(q1, 3, M1 * 3, M1,
                                                       q2, fps2, M2, 0.4f * 0.4f, nidx2);
  sa_stage2_kernel<<<NB * M2, 128, 0, stream>>>(x1, q1, q2, nidx2, w1b, w2b, w3b,
                                                sa2b[0], sa2s[0], sa2h[0],
                                                sa2b[1], sa2s[1], sa2h[1],
                                                sa2b[2], x2);

  // 4) SA3 global MLP + max
  sa_stage3_kernel<<<(NB * M2) / 16, 128, 0, stream>>>(x2, q2, w1c, w2c, w3c,
                                                       sa3b[0], sa3s[0], sa3h[0],
                                                       sa3b[1], sa3s[1], sa3h[1],
                                                       sa3b[2], gkeys);

  // 5) head
  head_kernel<<<NB, 512, 0, stream>>>(gkeys, hW[0], hb[0], hW[1], hb[1], hW[2], hb[2], out);
}
